// Decoder_51290499448871
// MI455X (gfx1250) — matrix operand order repaired and checked by reading, not compiled, not run on hardware
//
#include <hip/hip_runtime.h>
#include <math.h>

// ---------------------------------------------------------------------------
// Types for CDNA5 WMMA (gfx1250, wave32)
// ---------------------------------------------------------------------------
typedef __bf16 bf16;
typedef bf16  v16bf __attribute__((ext_vector_type(16)));
typedef float v8f   __attribute__((ext_vector_type(8)));

union V16U { v16bf v; bf16 e[16]; };
union V8U  { v8f   v; float e[8]; };

static __device__ __forceinline__ float sigmoidf_(float x) {
    return 1.0f / (1.0f + __expf(-x));
}

// Problem constants
#define PB 32
#define PL 120
#define PT 1000
#define PV 10000
#define PE 512
#define PH 512

// ---------------------------------------------------------------------------
// f32 -> bf16 conversion (one-time, for weights and encoder_output)
// ---------------------------------------------------------------------------
__global__ void cvt_f32_bf16(const float* __restrict__ src, bf16* __restrict__ dst, int n) {
    int i = blockIdx.x * blockDim.x + threadIdx.x;
    if (i < n) dst[i] = (bf16)src[i];
}

__global__ void zero_f32(float* __restrict__ p, int n) {
    int i = blockIdx.x * blockDim.x + threadIdx.x;
    if (i < n) p[i] = 0.0f;
}

// ---------------------------------------------------------------------------
// Embedding gather directly to bf16 (emb only feeds a bf16 WMMA GEMM)
// ---------------------------------------------------------------------------
__global__ void gather_emb(const int* __restrict__ idx, const float* __restrict__ table,
                           bf16* __restrict__ emb) {
    size_t i = (size_t)blockIdx.x * blockDim.x + threadIdx.x;
    if (i >= (size_t)PB * PL * PE) return;
    int m = (int)(i / PE);
    int e = (int)(i % PE);
    emb[i] = (bf16)table[(size_t)idx[m] * PE + e];
}

// ---------------------------------------------------------------------------
// WMMA fragment loaders (layouts used by this code, wave32; A/B use contiguous k, not the ISA table order):
//   A 16x32: lane 0-15 row=lane, K = half*16 + e
//   B 32x16: N=lane&15, K = half*16 + e
//   C 16x16: row = v + 8*half, col = lane&15
// ---------------------------------------------------------------------------
static __device__ __forceinline__ v16bf load_a_frag(const bf16* __restrict__ Ap,
                                                    int k0, int half) {
    V16U a;
    #pragma unroll
    for (int e = 0; e < 16; ++e) {
        a.e[e] = Ap[k0 + half * 16 + e];
    }
    return a.v;
}

static __device__ __forceinline__ v16bf load_b_frag(const bf16* __restrict__ Bp, int k0) {
    V16U b;
    #pragma unroll
    for (int e = 0; e < 16; ++e) b.e[e] = Bp[k0 + e];
    return b.v;
}

// ---------------------------------------------------------------------------
// WMMA GEMM, 4 N-tiles per wave: C[M,N] = A[M,K](bf16) * W[N,K](bf16)^T + bias
// One wave (32 threads) per 16x64 output strip; A fragment reused 4x.
// Requires M%16==0, K%32==0, and grid.x*64 <= N with N%16==0.
// ---------------------------------------------------------------------------
__global__ __launch_bounds__(32) void gemm_bf16_wmma4(
    const bf16* __restrict__ A, const bf16* __restrict__ Bw,
    const float* __restrict__ bias, float* __restrict__ C,
    int M, int N, int K)
{
    const int lane = threadIdx.x & 31;
    const int half = lane >> 4;
    const int l16  = lane & 15;
    const int nt0  = blockIdx.x * 4;
    const int mt   = blockIdx.y;

    const bf16* Ap = A  + (size_t)(mt * 16 + l16) * K;
    const bf16* B0 = Bw + (size_t)(nt0 * 16 + l16) * K + half * 16;
    const bf16* B1 = B0 + (size_t)16 * K;
    const bf16* B2 = B0 + (size_t)32 * K;
    const bf16* B3 = B0 + (size_t)48 * K;

    v8f acc0 = {}, acc1 = {}, acc2 = {}, acc3 = {};
    for (int k0 = 0; k0 < K; k0 += 32) {
        v16bf a = load_a_frag(Ap, k0, half);
        __builtin_prefetch(B0 + k0 + 64, 0, 1);   // global_prefetch_b8
        __builtin_prefetch(B2 + k0 + 64, 0, 1);
        v16bf b0 = load_b_frag(B0, k0);
        v16bf b1 = load_b_frag(B1, k0);
        v16bf b2 = load_b_frag(B2, k0);
        v16bf b3 = load_b_frag(B3, k0);
        acc0 = __builtin_amdgcn_wmma_f32_16x16x32_bf16(false, a, false, b0, (short)0, acc0, false, false);
        acc1 = __builtin_amdgcn_wmma_f32_16x16x32_bf16(false, a, false, b1, (short)0, acc1, false, false);
        acc2 = __builtin_amdgcn_wmma_f32_16x16x32_bf16(false, a, false, b2, (short)0, acc2, false, false);
        acc3 = __builtin_amdgcn_wmma_f32_16x16x32_bf16(false, a, false, b3, (short)0, acc3, false, false);
    }

    V8U o0, o1, o2, o3; o0.v = acc0; o1.v = acc1; o2.v = acc2; o3.v = acc3;
    #pragma unroll
    for (int v = 0; v < 8; ++v) {
        int row = mt * 16 + v + 8 * half;
        float* Cr = C + (size_t)row * N;
        int c0 = nt0 * 16 + l16;
        Cr[c0]      = o0.e[v] + (bias ? bias[c0]      : 0.0f);
        Cr[c0 + 16] = o1.e[v] + (bias ? bias[c0 + 16] : 0.0f);
        Cr[c0 + 32] = o2.e[v] + (bias ? bias[c0 + 32] : 0.0f);
        Cr[c0 + 48] = o3.e[v] + (bias ? bias[c0 + 48] : 0.0f);
    }
}

// Single-tile cleanup GEMM (for N not divisible by 64); ntOff = tile offset.
__global__ __launch_bounds__(32) void gemm_bf16_wmma1(
    const bf16* __restrict__ A, const bf16* __restrict__ Bw,
    const float* __restrict__ bias, float* __restrict__ C,
    int M, int N, int K, int ntOff)
{
    const int lane = threadIdx.x & 31;
    const int half = lane >> 4;
    const int l16  = lane & 15;
    const int nt = blockIdx.x + ntOff;
    const int mt = blockIdx.y;
    const bf16* Ap = A  + (size_t)(mt * 16 + l16) * K;
    const bf16* Bp = Bw + (size_t)(nt * 16 + l16) * K + half * 16;

    v8f acc = {};
    for (int k0 = 0; k0 < K; k0 += 32) {
        v16bf a = load_a_frag(Ap, k0, half);
        v16bf b = load_b_frag(Bp, k0);
        acc = __builtin_amdgcn_wmma_f32_16x16x32_bf16(false, a, false, b, (short)0, acc, false, false);
    }
    const float bv = bias ? bias[nt * 16 + l16] : 0.0f;
    V8U o; o.v = acc;
    #pragma unroll
    for (int v = 0; v < 8; ++v) {
        int row = mt * 16 + v + 8 * half;
        C[(size_t)row * N + nt * 16 + l16] = o.e[v] + bv;
    }
}

// ---------------------------------------------------------------------------
// Fused LSTM-from-zero-state layer: gates = A @ W^T + b_ih + b_hh (gate order
// i,f,g,o; f unused since c0=0).  h = sigmoid(o) * tanh(sigmoid(i)*tanh(g)).
// W is [4H, K] bf16. One wave computes the i/g/o 16x16 gate tiles for one
// 16x16 tile of h (3 WMMAs per A fragment); writes bf16 (+ optional f32).
// ---------------------------------------------------------------------------
__global__ __launch_bounds__(32) void lstm_gemm_wmma(
    const bf16* __restrict__ A, const bf16* __restrict__ W,
    const float* __restrict__ b_ih, const float* __restrict__ b_hh,
    bf16* __restrict__ HoutB, float* __restrict__ HoutF, int M, int K)
{
    const int lane = threadIdx.x & 31;
    const int half = lane >> 4;
    const int l16  = lane & 15;
    const int nt = blockIdx.x, mt = blockIdx.y;
    const int n  = nt * 16 + l16;                 // output column in [0,512)
    const bf16* Ap = A + (size_t)(mt * 16 + l16) * K;
    const bf16* Bi = W + (size_t)(n)        * K + half * 16;   // i gate rows
    const bf16* Bg = W + (size_t)(1024 + n) * K + half * 16;   // g gate rows
    const bf16* Bo = W + (size_t)(1536 + n) * K + half * 16;   // o gate rows

    v8f ai = {}, ag = {}, ao = {};
    for (int k0 = 0; k0 < K; k0 += 32) {
        v16bf a  = load_a_frag(Ap, k0, half);
        __builtin_prefetch(Bg + k0 + 64, 0, 1);
        v16bf bi = load_b_frag(Bi, k0);
        v16bf bg = load_b_frag(Bg, k0);
        v16bf bo = load_b_frag(Bo, k0);
        ai = __builtin_amdgcn_wmma_f32_16x16x32_bf16(false, a, false, bi, (short)0, ai, false, false);
        ag = __builtin_amdgcn_wmma_f32_16x16x32_bf16(false, a, false, bg, (short)0, ag, false, false);
        ao = __builtin_amdgcn_wmma_f32_16x16x32_bf16(false, a, false, bo, (short)0, ao, false, false);
    }
    const float bib = b_ih[n]        + b_hh[n];
    const float bgb = b_ih[1024 + n] + b_hh[1024 + n];
    const float bob = b_ih[1536 + n] + b_hh[1536 + n];
    V8U oi, og, oo; oi.v = ai; og.v = ag; oo.v = ao;
    #pragma unroll
    for (int v = 0; v < 8; ++v) {
        int row = mt * 16 + v + 8 * half;
        float iv = sigmoidf_(oi.e[v] + bib);
        float gv = tanhf(og.e[v] + bgb);
        float ov = sigmoidf_(oo.e[v] + bob);
        float h  = ov * tanhf(iv * gv);
        HoutB[(size_t)row * PH + n] = (bf16)h;
        if (HoutF) HoutF[(size_t)row * PH + n] = h;
    }
}

// ---------------------------------------------------------------------------
// Attention energies + score: one wave per (b,t).  qproj already contains
// attn_bias (folded into the q_proj GEMM bias epilogue).
// score[b,t] = score_w . tanh(qp[b,l,:] + vproj[b,t,:] + conv(attn_prev)) + score_b
// ---------------------------------------------------------------------------
__global__ __launch_bounds__(256) void attn_score(
    const float* __restrict__ qproj, int l,
    const float* __restrict__ vproj, const float* __restrict__ attn_prev,
    const float* __restrict__ conv_w, const float* __restrict__ conv_b,
    const float* __restrict__ score_w,
    const float* __restrict__ score_b, float* __restrict__ score)
{
    int gid = blockIdx.x * blockDim.x + threadIdx.x;
    int wid = gid >> 5;
    int lane = gid & 31;
    if (wid >= PB * PT) return;
    int b = wid / PT, t = wid % PT;

    float apm = (t > 0)      ? attn_prev[b * PT + t - 1] : 0.0f;
    float ap0 =                attn_prev[b * PT + t];
    float app = (t < PT - 1) ? attn_prev[b * PT + t + 1] : 0.0f;

    const float* qp = qproj + ((size_t)b * PL + l) * PH;
    const float* vp = vproj + ((size_t)b * PT + t) * PH;

    float s = 0.0f;
    const int h0 = lane * 16;
    #pragma unroll 4
    for (int h = h0; h < h0 + 16; ++h) {
        float conv = fmaf(conv_w[h * 3 + 0], apm,
                     fmaf(conv_w[h * 3 + 1], ap0,
                     fmaf(conv_w[h * 3 + 2], app, conv_b[h])));
        float e = tanhf(qp[h] + vp[h] + conv);
        s = fmaf(score_w[h], e, s);
    }
    #pragma unroll
    for (int off = 16; off > 0; off >>= 1) s += __shfl_xor(s, off, 32);
    if (lane == 0) score[b * PT + t] = s + score_b[0];
}

// ---------------------------------------------------------------------------
// Softmax over T + context GEMV, one block of 256 per batch row.
// ---------------------------------------------------------------------------
__global__ __launch_bounds__(256) void softmax_ctx(
    const float* __restrict__ score, const float* __restrict__ enc,
    float* __restrict__ attn_out, float* __restrict__ context)
{
    __shared__ float sAttn[PT];
    __shared__ float red[256];
    int b = blockIdx.x, tid = threadIdx.x;

    float m = -3.4e38f;
    for (int t = tid; t < PT; t += 256) { float v = score[b * PT + t]; sAttn[t] = v; m = fmaxf(m, v); }
    red[tid] = m; __syncthreads();
    for (int s = 128; s > 0; s >>= 1) { if (tid < s) red[tid] = fmaxf(red[tid], red[tid + s]); __syncthreads(); }
    m = red[0]; __syncthreads();

    float sum = 0.0f;
    for (int t = tid; t < PT; t += 256) { float p = __expf(sAttn[t] - m); sAttn[t] = p; sum += p; }
    red[tid] = sum; __syncthreads();
    for (int s = 128; s > 0; s >>= 1) { if (tid < s) red[tid] += red[tid + s]; __syncthreads(); }
    float inv = 1.0f / red[0]; __syncthreads();

    for (int t = tid; t < PT; t += 256) { sAttn[t] *= inv; attn_out[b * PT + t] = sAttn[t]; }
    __syncthreads();

    for (int h = tid; h < PH; h += 256) {
        const float* ep = enc + (size_t)b * PT * PH + h;
        float c = 0.0f;
        for (int t = 0; t < PT; ++t) c = fmaf(sAttn[t], ep[(size_t)t * PH], c);
        context[b * PH + h] = c;
    }
}

// ---------------------------------------------------------------------------
// x = tanh([context, q_t]) -> (B, 2H) bf16 (feeds the fc WMMA GEMM)
// ---------------------------------------------------------------------------
__global__ void cat_tanh(const float* __restrict__ context, const float* __restrict__ q,
                         int l, bf16* __restrict__ x)
{
    int i = blockIdx.x * blockDim.x + threadIdx.x;
    if (i >= PB * 2 * PH) return;
    int b = i >> 10;          // 2H = 1024
    int c = i & 1023;
    float v = (c < PH) ? context[b * PH + c]
                       : q[((size_t)b * PL + l) * PH + (c - PH)];
    x[i] = (bf16)tanhf(v);
}

// ---------------------------------------------------------------------------
// log_softmax over V, write to out[b, l, :]
// ---------------------------------------------------------------------------
__global__ __launch_bounds__(256) void log_softmax_out(
    const float* __restrict__ logits, int l, float* __restrict__ out)
{
    __shared__ float red[256];
    int b = blockIdx.x, tid = threadIdx.x;
    const float* lg = logits + (size_t)b * PV;

    float m = -3.4e38f;
    for (int v = tid; v < PV; v += 256) m = fmaxf(m, lg[v]);
    red[tid] = m; __syncthreads();
    for (int s = 128; s > 0; s >>= 1) { if (tid < s) red[tid] = fmaxf(red[tid], red[tid + s]); __syncthreads(); }
    m = red[0]; __syncthreads();

    float sum = 0.0f;
    for (int v = tid; v < PV; v += 256) sum += __expf(lg[v] - m);
    red[tid] = sum; __syncthreads();
    for (int s = 128; s > 0; s >>= 1) { if (tid < s) red[tid] += red[tid + s]; __syncthreads(); }
    float lse = m + __logf(red[0]);

    float* op = out + ((size_t)b * PL + l) * PV;
    for (int v = tid; v < PV; v += 256) op[v] = lg[v] - lse;
}

// ---------------------------------------------------------------------------
// Host orchestration
// ---------------------------------------------------------------------------
extern "C" void kernel_launch(void* const* d_in, const int* in_sizes, int n_in,
                              void* d_out, int out_size, void* d_ws, size_t ws_size,
                              hipStream_t stream)
{
    (void)in_sizes; (void)n_in; (void)out_size; (void)ws_size;
    const int B = PB, L = PL, T = PT, V = PV, E = PE, H = PH;
    const int BL = B * L, G = 4 * H;

    const int*   inputs    = (const int*)  d_in[0];
    const float* enc       = (const float*)d_in[1];
    const float* embTab    = (const float*)d_in[2];
    const float* W_ih0     = (const float*)d_in[3];
    const float* b_ih0     = (const float*)d_in[4];
    const float* b_hh0     = (const float*)d_in[5];
    const float* W_ih1     = (const float*)d_in[6];
    const float* b_ih1     = (const float*)d_in[7];
    const float* b_hh1     = (const float*)d_in[8];
    const float* conv_w    = (const float*)d_in[9];
    const float* conv_b    = (const float*)d_in[10];
    const float* Wq        = (const float*)d_in[11];
    const float* Wv        = (const float*)d_in[12];
    const float* attn_bias = (const float*)d_in[13];
    const float* score_w   = (const float*)d_in[14];
    const float* score_b   = (const float*)d_in[15];
    const float* fc_w      = (const float*)d_in[16];
    const float* fc_b      = (const float*)d_in[17];
    float* out = (float*)d_out;

    char* base = (char*)d_ws;
    size_t off = 0;
    auto alloc = [&](size_t bytes) -> void* {
        off = (off + 255) & ~(size_t)255;
        void* p = base + off; off += bytes; return p;
    };
    bf16*  Wih0B = (bf16*) alloc((size_t)G * E * 2);
    bf16*  Wih1B = (bf16*) alloc((size_t)G * H * 2);
    bf16*  WqB   = (bf16*) alloc((size_t)H * H * 2);
    bf16*  WvB   = (bf16*) alloc((size_t)H * H * 2);
    bf16*  fcwB  = (bf16*) alloc((size_t)V * 2 * H * 2);
    bf16*  embB  = (bf16*) alloc((size_t)BL * E * 2);
    bf16*  h0B   = (bf16*) alloc((size_t)BL * H * 2);
    bf16*  qB    = (bf16*) alloc((size_t)BL * H * 2);
    bf16*  encB  = (bf16*) alloc((size_t)B * T * H * 2);
    float* qF    = (float*)alloc((size_t)BL * H * 4);
    float* qproj = (float*)alloc((size_t)BL * H * 4);
    float* vproj = (float*)alloc((size_t)B * T * H * 4);
    float* attnA = (float*)alloc((size_t)B * T * 4);
    float* attnB = (float*)alloc((size_t)B * T * 4);
    float* score = (float*)alloc((size_t)B * T * 4);
    float* ctx   = (float*)alloc((size_t)B * H * 4);
    bf16*  xcatB = (bf16*) alloc((size_t)B * 2 * H * 2);
    float* logit = (float*)alloc((size_t)B * V * 4);

    auto cdiv = [](size_t a, size_t b) -> unsigned { return (unsigned)((a + b - 1) / b); };

    // --- prologue: one-time bf16 conversions ---
    cvt_f32_bf16<<<cdiv((size_t)G * E, 256), 256, 0, stream>>>(W_ih0, Wih0B, G * E);
    cvt_f32_bf16<<<cdiv((size_t)G * H, 256), 256, 0, stream>>>(W_ih1, Wih1B, G * H);
    cvt_f32_bf16<<<cdiv((size_t)H * H, 256), 256, 0, stream>>>(Wq, WqB, H * H);
    cvt_f32_bf16<<<cdiv((size_t)H * H, 256), 256, 0, stream>>>(Wv, WvB, H * H);
    cvt_f32_bf16<<<cdiv((size_t)V * 2 * H, 256), 256, 0, stream>>>(fc_w, fcwB, V * 2 * H);
    cvt_f32_bf16<<<cdiv((size_t)B * T * H, 256), 256, 0, stream>>>(enc, encB, B * T * H);

    // --- prologue: embeddings + batched projections (all L steps at once) ---
    gather_emb<<<cdiv((size_t)BL * E, 256), 256, 0, stream>>>(inputs, embTab, embB);
    lstm_gemm_wmma<<<dim3(H / 16, BL / 16), 32, 0, stream>>>(embB, Wih0B, b_ih0, b_hh0, h0B, nullptr, BL, E);
    lstm_gemm_wmma<<<dim3(H / 16, BL / 16), 32, 0, stream>>>(h0B, Wih1B, b_ih1, b_hh1, qB, qF, BL, H);
    // q_proj with attn_bias folded into the GEMM epilogue
    gemm_bf16_wmma4<<<dim3(H / 64, BL / 16), 32, 0, stream>>>(qB, WqB, attn_bias, qproj, BL, H, H);
    gemm_bf16_wmma4<<<dim3(H / 64, (B * T) / 16), 32, 0, stream>>>(encB, WvB, nullptr, vproj, B * T, H, H);
    zero_f32<<<cdiv((size_t)B * T, 256), 256, 0, stream>>>(attnA, B * T);

    // --- sequential attention scan over L decode steps ---
    float* ap = attnA;
    float* an = attnB;
    const int fcTiles4 = (V / 16) / 4;           // 156 four-tile strips
    const int fcRem    = (V / 16) - fcTiles4 * 4; // 1 leftover tile
    for (int l = 0; l < L; ++l) {
        attn_score<<<cdiv((size_t)B * T * 32, 256), 256, 0, stream>>>(
            qproj, l, vproj, ap, conv_w, conv_b, score_w, score_b, score);
        softmax_ctx<<<B, 256, 0, stream>>>(score, enc, an, ctx);
        cat_tanh<<<cdiv((size_t)B * 2 * H, 256), 256, 0, stream>>>(ctx, qF, l, xcatB);
        gemm_bf16_wmma4<<<dim3(fcTiles4, B / 16), 32, 0, stream>>>(xcatB, fcwB, fc_b, logit, B, V, 2 * H);
        if (fcRem > 0)
            gemm_bf16_wmma1<<<dim3(fcRem, B / 16), 32, 0, stream>>>(xcatB, fcwB, fc_b, logit, B, V, 2 * H, fcTiles4 * 4);
        log_softmax_out<<<B, 256, 0, stream>>>(logit, l, out);
        float* tmp = ap; ap = an; an = tmp;
    }
}
